// Conv2dP4P4_74388833566913
// MI455X (gfx1250) — matrix operand order repaired and checked by reading, not compiled, not run on hardware
//
#include <hip/hip_runtime.h>

typedef _Float16 h16;
typedef __attribute__((ext_vector_type(16))) _Float16     v16h;
typedef __attribute__((ext_vector_type(8)))  float        v8f;
typedef __attribute__((ext_vector_type(4)))  unsigned int v4u;

// Problem constants
#define NB   16
#define NC   64
#define NH   128
#define NW   128
#define HO   126          // 128 - 3 + 1
#define PLANE (HO*HO)     // 15876

// LDS layout
#define PCELL 144                      // bytes per (h,w) cell: 64 f16 + 8 f16 pad (bank swizzle)
#define PATCH_BYTES (34*34*PCELL)      // 166464
#define WLDS_OFF    PATCH_BYTES
#define WLDS_BYTES  (9*16*PCELL)       // 20736
#define SMEM_BYTES  (PATCH_BYTES + WLDS_BYTES)  // 187200 <= 320KB/WGP

// ---------------------------------------------------------------------------
// Prep: weff_rot[i][m][pos][ic] = (f16) sum_j w[4m+j][ic][a'][b']
// tap permutation per rotation i (conv-of-rotated-image == rotated-conv with
// permuted taps):  i=0:(a,b)  i=1:(b,2-a)  i=2:(2-a,2-b)  i=3:(2-b,a)
// ---------------------------------------------------------------------------
__global__ __launch_bounds__(256) void p4_prep_weights(const float* __restrict__ w,
                                                       h16* __restrict__ wp) {
    int t = blockIdx.x * 256 + threadIdx.x;
    if (t >= 4 * 64 * 9 * 64) return;
    int ic  = t & 63;
    int q   = t >> 6;              // (i*64+m)*9 + pos
    int pos = q % 9;
    int m   = (q / 9) & 63;
    int i   = t / (64 * 9 * 64);
    int a = pos / 3, b = pos % 3;
    int ap, bp;
    if      (i == 0) { ap = a;     bp = b;     }
    else if (i == 1) { ap = b;     bp = 2 - a; }
    else if (i == 2) { ap = 2 - a; bp = 2 - b; }
    else             { ap = 2 - b; bp = a;     }
    float s = 0.f;
#pragma unroll
    for (int j = 0; j < 4; ++j)
        s += w[(((4 * m + j) * 64 + ic) * 3 + ap) * 3 + bp];
    wp[t] = (h16)s;
}

// ---------------------------------------------------------------------------
// Main: implicit-GEMM p4 conv.
// Block = (i, b, mtile(16 out-ch = one p4 group block), 32x32 spatial tile).
// C'[u,v] = sum_{a,b,ic} X[b, ic, ge, u+a, v+b] * Wrot[i][m][a,b,ic],
// ge = (mtile + i) & 3.  Output written with per-i rotation remap.
// ---------------------------------------------------------------------------
__global__ __launch_bounds__(256) void p4_conv_kernel(const float* __restrict__ x,
                                                      const h16* __restrict__ wp,
                                                      float* __restrict__ out) {
    extern __shared__ char smem[];
    const int tid = threadIdx.x;

    // bid -> (i, b, mtile, ty, tx); consecutive blocks share (i,b) for L2 reuse
    int bid   = blockIdx.x;
    int tx    = bid & 3;
    int ty    = (bid >> 2) & 3;
    int mtile = (bid >> 4) & 3;
    int bb    = (bid >> 6) & 15;
    int rot   = bid >> 10;
    int u0 = ty * 32, v0 = tx * 32;
    int ge = (mtile + rot) & 3;

    // ---- load rotated/folded weight slice (16 m x 9 pos x 64 ic, f16) ----
    const h16* wsrc = wp + (rot * 64 + mtile * 16) * 576;
    for (int fl = tid; fl < 16 * 576; fl += 256) {
        int m_l = fl / 576;
        int rem = fl - m_l * 576;
        int pos = rem >> 6;
        int ic  = rem & 63;
        *(h16*)(smem + WLDS_OFF + (pos * 16 + m_l) * PCELL + ic * 2) = wsrc[fl];
    }

    // ---- load 34x34x64 input patch (fp32 -> f16), cells of 144B ----
    const float* xs = x + (size_t)(bb * 256 + ge) * (NH * NW);   // (b*C + ic)*G + ge, ic stride = 4*H*W
    for (int fl = tid; fl < 64 * 34 * 34; fl += 256) {
        int ic  = fl / 1156;
        int rem = fl - ic * 1156;
        int hh  = rem / 34;
        int ww  = rem - hh * 34;
        int sh = u0 + hh; if (sh > 127) sh = 127;   // clamped rows feed only masked outputs
        int sw = v0 + ww; if (sw > 127) sw = 127;
        float v = xs[(size_t)ic * (4 * NH * NW) + sh * NW + sw];
        *(h16*)(smem + (hh * 34 + ww) * PCELL + ic * 2) = (h16)v;
    }
    __syncthreads();

    // ---- implicit GEMM: 18 K-steps of v_wmma_f32_16x16x32_f16, 8 N-tiles/wave ----
    const int lane = tid & 31;
    const int wv   = tid >> 5;
    const int nl   = lane & 15;
    const int hi   = lane >> 4;

    v8f acc[8];
#pragma unroll
    for (int t = 0; t < 8; ++t) acc[t] = (v8f){};

    int tbase[8];
#pragma unroll
    for (int t = 0; t < 8; ++t) {
        int urow = wv * 4 + (t & 3);
        int dx   = ((t >> 2) * 16) + nl;
        tbase[t] = (urow * 34 + dx) * PCELL + hi * 32;
    }
    const int abase = WLDS_OFF + nl * PCELL + hi * 32;

    union Frag { v4u q[2]; v16h v; };

#pragma unroll
    for (int ks = 0; ks < 18; ++ks) {
        const int pos    = ks >> 1;
        const int icb2   = (ks & 1) * 64;                       // ic byte offset (0 / 64)
        const int posoff = (((pos / 3) * 34) + (pos % 3)) * PCELL;

        Frag af;
        const char* ap = smem + abase + pos * (16 * PCELL) + icb2;
        af.q[0] = *(const v4u*)(ap);
        af.q[1] = *(const v4u*)(ap + 16);

#pragma unroll
        for (int t = 0; t < 8; ++t) {
            Frag bf;
            const char* bp = smem + tbase[t] + posoff + icb2;
            bf.q[0] = *(const v4u*)(bp);
            bf.q[1] = *(const v4u*)(bp + 16);
            acc[t] = __builtin_amdgcn_wmma_f32_16x16x32_f16(
                false, af.v, false, bf.v, (short)0, acc[t], false, false);
        }
    }
    __syncthreads();   // all patch reads done before aliasing LDS with C' tile

    // ---- accumulators -> LDS C' tile [16][32][33] f32 (33-pad: conflict-free transpose reads)
    float* ctile = (float*)smem;
#pragma unroll
    for (int t = 0; t < 8; ++t) {
        int urow = wv * 4 + (t & 3);
        int dv   = ((t >> 2) * 16) + nl;
#pragma unroll
        for (int r = 0; r < 8; ++r)
            ctile[(r + hi * 8) * (32 * 33) + urow * 33 + dv] = acc[t][r];
    }
    __syncthreads();

    // ---- rotation-remapped, coalesced global stores ----
    // out[b, m, i, y, x];  i=0: (y,x)=(u,v)  i=1: C'[125-x, y]
    //                      i=2: C'[125-y,125-x]  i=3: C'[x, 125-y]
    const size_t obase = ((size_t)(bb * 64 + mtile * 16) * 4 + rot) * PLANE;
    for (int it = 0; it < 64; ++it) {
        int flat = it * 256 + tid;
        int m_l  = flat >> 10;
        int rem  = flat & 1023;
        int line = rem >> 5;
        int col  = rem & 31;
        float val; int yy, xx; bool ok;
        if (rot == 0) {
            val = ctile[m_l * 1056 + line * 33 + col];
            yy = u0 + line; xx = v0 + col;
            ok = (yy < HO) && (xx < HO);
        } else if (rot == 1) {
            val = ctile[m_l * 1056 + col * 33 + line];
            yy = v0 + line; xx = 125 - (u0 + col);
            ok = (yy < HO) && (u0 + col < HO);
        } else if (rot == 2) {
            val = ctile[m_l * 1056 + line * 33 + col];
            yy = 125 - (u0 + line); xx = 125 - (v0 + col);
            ok = (u0 + line < HO) && (v0 + col < HO);
        } else {
            val = ctile[m_l * 1056 + col * 33 + line];
            yy = 125 - (v0 + line); xx = u0 + col;
            ok = (xx < HO) && (v0 + line < HO);
        }
        if (ok)
            out[obase + (size_t)m_l * 4 * PLANE + yy * HO + xx] = val;
    }
}

// ---------------------------------------------------------------------------
extern "C" void kernel_launch(void* const* d_in, const int* in_sizes, int n_in,
                              void* d_out, int out_size, void* d_ws, size_t ws_size,
                              hipStream_t stream) {
    (void)in_sizes; (void)n_in; (void)out_size; (void)ws_size;
    const float* x = (const float*)d_in[0];
    const float* w = (const float*)d_in[1];
    float* out = (float*)d_out;
    h16* wp = (h16*)d_ws;   // 4*64*9*64 f16 = 294,912 B of scratch

    p4_prep_weights<<<576, 256, 0, stream>>>(w, wp);
    p4_conv_kernel<<<4096, 256, SMEM_BYTES, stream>>>(x, wp, out);
}
